// VectorQuantizer_77455440216578
// MI455X (gfx1250) — matrix operand order repaired and checked by reading, not compiled, not run on hardware
//
#include <hip/hip_runtime.h>
#include <hip/hip_bf16.h>

typedef __attribute__((ext_vector_type(16))) __bf16 v16bf;
typedef __attribute__((ext_vector_type(8)))  float  v8f;
typedef unsigned short u16;
typedef unsigned long long u64;

#define DIM     256
#define NCODES  2048
#define NROWS   65536     // B*T = 16*4096
#define MTILE   128
#define NCHUNK  64

__device__ __forceinline__ u16 f2bf(float x) {
    unsigned int u = __float_as_uint(x);
    u += 0x7FFFu + ((u >> 16) & 1u);   // round-to-nearest-even
    return (u16)(u >> 16);
}

// ---------------------------------------------------------------------------
// 0) zero the per-launch accumulators (usage histogram + loss scalar)
// ---------------------------------------------------------------------------
__global__ void vq_init_kernel(float* __restrict__ usage, float* __restrict__ lossAcc) {
    int t = blockIdx.x * 256 + threadIdx.x;
    if (t < NCODES) usage[t] = 0.0f;
    if (t == 0) lossAcc[0] = 0.0f;
}

// ---------------------------------------------------------------------------
// 1) codebook f32 -> bf16 + ||W||^2 per code.  One wave per code row.
// ---------------------------------------------------------------------------
__global__ __launch_bounds__(256) void vq_prep_kernel(
    const float* __restrict__ cb, u16* __restrict__ cbb, float* __restrict__ cbsq)
{
    int wave = threadIdx.x >> 5;
    int lane = threadIdx.x & 31;
    int row  = blockIdx.x * 8 + wave;
    const float* src = cb + (size_t)row * DIM;
    u16* dst = cbb + (size_t)row * DIM;
    float ss = 0.0f;
#pragma unroll
    for (int i = 0; i < 8; ++i) {
        float v = src[lane + i * 32];
        ss += v * v;
        dst[lane + i * 32] = f2bf(v);
    }
#pragma unroll
    for (int off = 16; off > 0; off >>= 1) ss += __shfl_xor(ss, off, 32);
    if (lane == 0) cbsq[row] = ss;
}

// ---------------------------------------------------------------------------
// 2) fused bf16-WMMA GEMM + argmin, 2M x 2N register blocking per wave,
//    with explicit double-buffering of B fragments (load k+1 before WMMA k).
//    block = 256 threads (8 waves), owns MTILE=128 rows of z.
//    wave = (waveM in 0..3) x (waveN in 0..1).
// ---------------------------------------------------------------------------
__global__ __launch_bounds__(256) void vq_gemm_argmin_kernel(
    const float* __restrict__ z, const u16* __restrict__ cbb,
    const float* __restrict__ cbsq, int* __restrict__ idx_out,
    float* __restrict__ idxf_out, float* __restrict__ usage)
{
    __shared__ u16 Abuf[MTILE * DIM];          // 64 KB bf16 A-tile (reused later)
    const int tid     = threadIdx.x;
    const int block_m = blockIdx.x * MTILE;

    // ---- stage + convert A tile (f32 -> bf16) into LDS ----
    {
        int r  = tid >> 1;
        int c0 = (tid & 1) * 128;
        const float4* zp = (const float4*)(z + (size_t)(block_m + r) * DIM + c0);
        u64* dst = (u64*)&Abuf[r * DIM + c0];
#pragma unroll 8
        for (int i = 0; i < 32; ++i) {
            float4 f = zp[i];
            u64 pk =  (u64)f2bf(f.x)
                   | ((u64)f2bf(f.y) << 16)
                   | ((u64)f2bf(f.z) << 32)
                   | ((u64)f2bf(f.w) << 48);
            dst[i] = pk;
        }
    }
    __syncthreads();

    const int lane  = tid & 31;
    const int wave  = tid >> 5;
    const int waveM = wave >> 1;    // 0..3 : which 32-row slice
    const int waveN = wave & 1;     // 0..1 : which half of the N chunk
    const int lo    = lane & 15;    // column / row-in-tile selector
    const int hi    = lane >> 4;    // lane-half selector

    // ---- preload 16 A fragments (2 M-tiles x full K=256), 16-bit A layout ----
    // lanes 0-15:  K 0..7 then 16..23 ; lanes 16-31: K 8..15 then 24..31
    v16bf afrag[2][8];
#pragma unroll
    for (int mt = 0; mt < 2; ++mt) {
        const u16* arow = &Abuf[(waveM * 32 + mt * 16 + lo) * DIM];
#pragma unroll
        for (int k = 0; k < 8; ++k) {
            const u16* p = arow + k * 32;
            uint4* aq = (uint4*)&afrag[mt][k];
            aq[0] = *(const uint4*)(p + hi * 8);
            aq[1] = *(const uint4*)(p + 16 + hi * 8);
        }
    }

    float best[2][8];
    int   bidx[2][8];
#pragma unroll
    for (int mt = 0; mt < 2; ++mt)
#pragma unroll
        for (int r = 0; r < 8; ++r) { best[mt][r] = 3.4e38f; bidx[mt][r] = 0; }

    // base pointer for this wave's two code columns (j selects +16 rows)
    const u16* bbase = cbb + (size_t)(waveN * 32 + lo) * DIM + hi * 8;

    // ---- N loop: 2048 codes in chunks of 64 (this wave sees 32 of each) ----
    for (int nb = 0; nb < NCODES; nb += NCHUNK) {
        if (nb + NCHUNK < NCODES)
            __builtin_prefetch(cbb + (size_t)(nb + NCHUNK + waveN * 32 + lo) * DIM, 0, 1);

        float csq[2];
#pragma unroll
        for (int j = 0; j < 2; ++j)
            csq[j] = cbsq[nb + (waveN * 2 + j) * 16 + lo];

        v8f c[2][2];
#pragma unroll
        for (int mt = 0; mt < 2; ++mt)
#pragma unroll
            for (int j = 0; j < 2; ++j)
#pragma unroll
                for (int r = 0; r < 8; ++r) c[mt][j][r] = 0.0f;

        const u16* bchunk = bbase + (size_t)nb * DIM;

        // ---- K loop with explicit double-buffered B fragments ----
        v16bf bbuf[2][2];                  // [parity][j]
#pragma unroll
        for (int j = 0; j < 2; ++j) {      // prologue: load k=0
            const u16* bp = bchunk + (size_t)j * 16 * DIM;
            uint4* bq = (uint4*)&bbuf[0][j];
            bq[0] = *(const uint4*)(bp);
            bq[1] = *(const uint4*)(bp + 16);
        }
#pragma unroll
        for (int k = 0; k < 8; ++k) {
            const int cur = k & 1;
            const int nxt = cur ^ 1;
            if (k < 7) {
#pragma unroll
                for (int j = 0; j < 2; ++j) {   // issue loads for k+1 first
                    const u16* bp = bchunk + (size_t)j * 16 * DIM + (k + 1) * 32;
                    uint4* bq = (uint4*)&bbuf[nxt][j];
                    bq[0] = *(const uint4*)(bp);
                    bq[1] = *(const uint4*)(bp + 16);
                }
            }
#pragma unroll
            for (int mt = 0; mt < 2; ++mt)
#pragma unroll
                for (int j = 0; j < 2; ++j)
                    c[mt][j] = __builtin_amdgcn_wmma_f32_16x16x32_bf16(
                        /*neg_a=*/false, afrag[mt][k], /*neg_b=*/false, bbuf[cur][j],
                        /*c_mod=*/(short)0, c[mt][j],
                        /*reuse_a=*/false, /*reuse_b=*/false);
        }

        // ---- fused distance + running argmin (z_sq drops out of argmin) ----
#pragma unroll
        for (int j = 0; j < 2; ++j) {
            int n = nb + (waveN * 2 + j) * 16 + lo;
#pragma unroll
            for (int mt = 0; mt < 2; ++mt)
#pragma unroll
                for (int r = 0; r < 8; ++r) {
                    float s = csq[j] - 2.0f * c[mt][j][r];
                    if (s < best[mt][r]) { best[mt][r] = s; bidx[mt][r] = n; }
                }
        }
    }

    // ---- reduce across the 16 lanes holding different N columns ----
    // C layout: lanes 0-15 hold M=r (vgpr slot r), lanes 16-31 hold M=8+r.
#pragma unroll
    for (int mt = 0; mt < 2; ++mt)
#pragma unroll
        for (int r = 0; r < 8; ++r)
#pragma unroll
            for (int off = 1; off < 16; off <<= 1) {
                float ov = __shfl_xor(best[mt][r], off, 32);
                int   oi = __shfl_xor(bidx[mt][r], off, 32);
                if (ov < best[mt][r] || (ov == best[mt][r] && oi < bidx[mt][r])) {
                    best[mt][r] = ov; bidx[mt][r] = oi;
                }
            }

    // ---- cross-wave combine (waveN=0 vs waveN=1) via LDS (reuse Abuf) ----
    float* candv = (float*)&Abuf[0];       // [128][2] floats  (1 KB)
    int*   candi = (int*)&Abuf[2048];      // [128][2] ints    (1 KB, byte off 4096)
    __syncthreads();                       // done reading Abuf as A-tile
    if (lo == 0) {
#pragma unroll
        for (int mt = 0; mt < 2; ++mt)
#pragma unroll
            for (int r = 0; r < 8; ++r) {
                int m_local = waveM * 32 + mt * 16 + hi * 8 + r;
                candv[m_local * 2 + waveN] = best[mt][r];
                candi[m_local * 2 + waveN] = bidx[mt][r];
            }
    }
    __syncthreads();
    if (tid < MTILE) {
        float v0 = candv[tid * 2 + 0], v1 = candv[tid * 2 + 1];
        int   i0 = candi[tid * 2 + 0], i1 = candi[tid * 2 + 1];
        int   pick1 = (v1 < v0) || (v1 == v0 && i1 < i0);
        int   bi = pick1 ? i1 : i0;
        int   m  = block_m + tid;
        idx_out[m]  = bi;
        idxf_out[m] = (float)bi;
        atomicAdd(&usage[bi], 1.0f);
    }
}

// ---------------------------------------------------------------------------
// 3) gather zq (f32 codebook), write zq_st, accumulate commitment loss
// ---------------------------------------------------------------------------
__global__ __launch_bounds__(256) void vq_gather_kernel(
    const float* __restrict__ z, const float* __restrict__ cb,
    const int* __restrict__ idx, float* __restrict__ zq_out,
    float* __restrict__ lossAcc)
{
    __shared__ float red[8];
    const int t = threadIdx.x;
    const int base_row = blockIdx.x * 8;
    float acc = 0.0f;
#pragma unroll
    for (int p = 0; p < 8; ++p) {
        int row = base_row + p;
        int id  = idx[row];
        float q  = cb[(size_t)id * DIM + t];
        float zv = z[(size_t)row * DIM + t];
        zq_out[(size_t)row * DIM + t] = q;   // forward value of zq_st == zq
        float d = q - zv;
        acc += d * d;
    }
#pragma unroll
    for (int off = 16; off > 0; off >>= 1) acc += __shfl_xor(acc, off, 32);
    if ((t & 31) == 0) red[t >> 5] = acc;
    __syncthreads();
    if (t < 8) {
        float v = red[t];
#pragma unroll
        for (int off = 4; off > 0; off >>= 1) v += __shfl_xor(v, off, 32);
        if (t == 0) atomicAdd(lossAcc, v);
    }
}

// ---------------------------------------------------------------------------
// 4) entropy of usage histogram + final scalars
// ---------------------------------------------------------------------------
__global__ __launch_bounds__(256) void vq_final_kernel(
    const float* __restrict__ usage, const float* __restrict__ lossAcc,
    float* __restrict__ out_scal)
{
    __shared__ float red[8];
    const int t = threadIdx.x;
    float e = 0.0f;
    for (int i = t; i < NCODES; i += 256) {
        float p = usage[i] * (1.0f / (float)NROWS);
        e -= p * __logf(p + 1e-10f);
    }
#pragma unroll
    for (int off = 16; off > 0; off >>= 1) e += __shfl_xor(e, off, 32);
    if ((t & 31) == 0) red[t >> 5] = e;
    __syncthreads();
    if (t < 8) {
        float v = red[t];
#pragma unroll
        for (int off = 4; off > 0; off >>= 1) v += __shfl_xor(v, off, 32);
        if (t == 0) {
            out_scal[0] = lossAcc[0] * (1.0f / ((float)NROWS * (float)DIM));
            out_scal[1] = __logf((float)NCODES) - v;
        }
    }
}

// ---------------------------------------------------------------------------
extern "C" void kernel_launch(void* const* d_in, const int* in_sizes, int n_in,
                              void* d_out, int out_size, void* d_ws, size_t ws_size,
                              hipStream_t stream) {
    const float* z   = (const float*)d_in[0];   // [16,4096,256] f32
    const float* cbw = (const float*)d_in[1];   // [2048,256]    f32

    float* out  = (float*)d_out;
    float* zq   = out;                                  // 65536*256
    float* idxf = out + (size_t)NROWS * DIM;            // 65536
    float* scal = idxf + NROWS;                         // 2 scalars

    char* ws = (char*)d_ws;
    u16*   cbb     = (u16*)ws;                                   // 1 MB bf16 codebook
    float* cbsq    = (float*)(ws + 1048576);                     // 8 KB
    int*   idxi    = (int*)(ws + 1048576 + 8192);                // 256 KB
    float* usage   = (float*)(ws + 1048576 + 8192 + 262144);     // 8 KB
    float* lossAcc = (float*)(ws + 1048576 + 8192 + 262144 + 8192);

    vq_init_kernel<<<8, 256, 0, stream>>>(usage, lossAcc);
    vq_prep_kernel<<<NCODES / 8, 256, 0, stream>>>(cbw, cbb, cbsq);
    vq_gemm_argmin_kernel<<<NROWS / MTILE, 256, 0, stream>>>(z, cbb, cbsq, idxi, idxf, usage);
    vq_gather_kernel<<<NROWS / 8, 256, 0, stream>>>(z, cbw, idxi, zq, lossAcc);
    vq_final_kernel<<<1, 256, 0, stream>>>(usage, lossAcc, scal);
}
